// CorefDecoderMangoes_48979807043767
// MI455X (gfx1250) — matrix operand order repaired and checked by reading, not compiled, not run on hardware
//
#include <hip/hip_runtime.h>
#include <hip/hip_bf16.h>
#include <stdint.h>

#define C_CAND 30000
#define D_EMB  1556
#define D_PAD  1600      // 50 * 32: K-dim and tile dims padded for 64x64 blocks
#define K_TOP  1600
#define A_ANT  50
#define NWORDS 4096
#define NEG_BIG (-1e9f)
#define BM 4             // 4x4 register blocking: 64x64 output per wave
#define BN 4

typedef __attribute__((ext_vector_type(16))) __bf16 v16bf;
typedef __attribute__((ext_vector_type(8)))  float  v8f;

union ABfrag { v16bf v; uint4 q[2]; };

__device__ __forceinline__ unsigned short f2bf(float f) {
  unsigned int u = __float_as_uint(f);
  u += 0x7FFFu + ((u >> 16) & 1u);      // round-to-nearest-even
  return (unsigned short)(u >> 16);
}

// ---------------------------------------------------------------------------
// 1) Stable descending argsort of mention scores via rank-by-counting.
// ---------------------------------------------------------------------------
__global__ void rank_scores_kernel(const float* __restrict__ scores,
                                   int* __restrict__ order) {
  __shared__ float sh[256];
  int i = blockIdx.x * 256 + threadIdx.x;
  float mys = (i < C_CAND) ? scores[i] : 0.0f;
  int rank = 0;
  for (int base = 0; base < C_CAND; base += 256) {
    int j = base + threadIdx.x;
    sh[threadIdx.x] = (j < C_CAND) ? scores[j] : 0.0f;
    __syncthreads();
    int lim = min(256, C_CAND - base);
    for (int t = 0; t < lim; ++t) {
      float s = sh[t];
      int jj = base + t;
      rank += (s > mys) || (s == mys && jj < i);
    }
    __syncthreads();
  }
  if (i < C_CAND) order[rank] = i;
}

__global__ void gather_se_kernel(const int* __restrict__ order,
                                 const int* __restrict__ starts,
                                 const int* __restrict__ ends,
                                 int* __restrict__ s_sorted,
                                 int* __restrict__ e_sorted) {
  int t = blockIdx.x * 256 + threadIdx.x;
  if (t < C_CAND) {
    int c = order[t];
    s_sorted[t] = starts[c];
    e_sorted[t] = ends[c];
  }
}

// ---------------------------------------------------------------------------
// 2) Sequential greedy non-crossing selection: single wave32, maps in LDS.
// ---------------------------------------------------------------------------
__global__ void greedy_select_kernel(const int* __restrict__ s_sorted,
                                     const int* __restrict__ e_sorted,
                                     int* __restrict__ sel_flags,
                                     int* __restrict__ n_sel_out) {
  __shared__ int s2me[NWORDS];
  __shared__ int e2ms[NWORDS];
  __shared__ int sbuf[512];
  __shared__ int ebuf[512];
  int lane = threadIdx.x;  // block = 32
  for (int t = lane; t < NWORDS; t += 32) { s2me[t] = -1; e2ms[t] = -1; }
  __syncthreads();
  int count = 0;
  for (int base = 0; base < C_CAND; base += 512) {
    int lim = min(512, C_CAND - base);
    for (int t = lane; t < lim; t += 32) {
      sbuf[t] = s_sorted[base + t];
      ebuf[t] = e_sorted[base + t];
    }
    __syncthreads();
    for (int t = 0; t < lim; ++t) {
      int s = sbuf[t], e = ebuf[t];
      int tok = s + lane;           // width <= 30, one wave covers [s, e]
      int cross = 0;
      if (tok <= e) {
        int me = s2me[tok];
        int ms = e2ms[tok];
        cross = ((tok > s) && (me > e)) ||
                ((tok < e) && (ms >= 0) && (ms < s));
      }
      int any = __any(cross);
      int sel = (!any && count < K_TOP) ? 1 : 0;
      if (lane == 0) {
        sel_flags[base + t] = sel;
        if (sel) {
          if (e > s2me[s]) s2me[s] = e;
          int pm = e2ms[e];
          e2ms[e] = (pm < 0 || s < pm) ? s : pm;
        }
      }
      count += sel;
      __syncthreads();
    }
  }
  if (lane == 0) *n_sel_out = count;
}

// ---------------------------------------------------------------------------
// 3) Stable re-sort of selected spans by (start*NW + end); unselected -> MAX.
// ---------------------------------------------------------------------------
__global__ void rank_selected_kernel(const int* __restrict__ s_sorted,
                                     const int* __restrict__ e_sorted,
                                     const int* __restrict__ sel_flags,
                                     const int* __restrict__ order,
                                     int* __restrict__ cand_by_rank) {
  __shared__ int shk[256];
  int p = blockIdx.x * 256 + threadIdx.x;
  int myk = 0x7fffffff;
  if (p < C_CAND && sel_flags[p])
    myk = s_sorted[p] * NWORDS + e_sorted[p];
  int rank = 0;
  for (int base = 0; base < C_CAND; base += 256) {
    int q = base + threadIdx.x;
    int kq = 0x7fffffff;
    if (q < C_CAND && sel_flags[q]) kq = s_sorted[q] * NWORDS + e_sorted[q];
    shk[threadIdx.x] = kq;
    __syncthreads();
    int lim = min(256, C_CAND - base);
    for (int t = 0; t < lim; ++t) {
      int k2 = shk[t];
      int qq = base + t;
      rank += (k2 < myk) || (k2 == myk && qq < p);
    }
    __syncthreads();
  }
  if (p < C_CAND && rank < K_TOP) cand_by_rank[rank] = order[p];
}

__global__ void fill_top_kernel(const int* __restrict__ cand_by_rank,
                                const int* __restrict__ n_sel_ptr,
                                int* __restrict__ top,
                                float* __restrict__ out_top) {
  int r = blockIdx.x * 256 + threadIdx.x;
  if (r < K_TOP) {
    int n_sel = *n_sel_ptr;
    int v = (r < n_sel) ? cand_by_rank[r] : cand_by_rank[0];
    top[r] = v;
    out_top[r] = (float)v;   // output slot 0: top_span_indices
  }
}

// ---------------------------------------------------------------------------
// 4) bf16 staging: gather top embeddings (K-dim padded to D_PAD with zeros),
//    convert W_fast, grab top mention scores.
// ---------------------------------------------------------------------------
__global__ void gather_emb_kernel(const int* __restrict__ top,
                                  const float* __restrict__ emb,
                                  const float* __restrict__ scores,
                                  unsigned short* __restrict__ Ebf,
                                  float* __restrict__ ts_top) {
  int idx = blockIdx.x * 256 + threadIdx.x;
  if (idx >= K_TOP * D_PAD) return;
  int i = idx / D_PAD;
  int d = idx - i * D_PAD;
  int c = top[i];
  Ebf[idx] = (d < D_EMB) ? f2bf(emb[c * D_EMB + d]) : (unsigned short)0;
  if (d == 0) ts_top[i] = scores[c];
}

__global__ void conv_w_kernel(const float* __restrict__ W,
                              unsigned short* __restrict__ Wbf) {
  int idx = blockIdx.x * 256 + threadIdx.x;
  if (idx >= D_PAD * D_PAD) return;
  int j = idx / D_PAD;
  int d = idx - j * D_PAD;
  Wbf[idx] = (j < D_EMB && d < D_EMB) ? f2bf(W[j * D_EMB + d])
                                      : (unsigned short)0;
}

__global__ void bucket_scores_kernel(const float* __restrict__ emb_dist,
                                     const float* __restrict__ W_dist,
                                     const float* __restrict__ b_dist,
                                     float* __restrict__ bucket_scores) {
  int b = threadIdx.x;
  if (b < 10) {
    float acc = b_dist[0];
    for (int f = 0; f < 20; ++f) acc += emb_dist[b * 20 + f] * W_dist[f];
    bucket_scores[b] = acc;
  }
}

// ---------------------------------------------------------------------------
// 5) WMMA bf16 GEMM core: 4x4 register-blocked 64x64 output per wave.
//    Per K-step: 16 b128 loads feed 16 v_wmma.  K-loop kept at unroll(1) and
//    the GEMM kernels request full register budget (waves_per_eu(1)) so the
//    16 f32 accumulators (128 VGPRs) + 8 fragments stay in registers.
// ---------------------------------------------------------------------------
__device__ __forceinline__ void wmma_block(const unsigned short* __restrict__ A,
                                           const unsigned short* __restrict__ B,
                                           v8f acc[BM][BN]) {
  int lane = threadIdx.x & 31;
  int m    = lane & 15;
  int kbA  = (lane >> 4) << 3;    // 0 or 8
  int kbB  = (lane >> 4) << 3;    // 0 or 8
  const unsigned short* ap = A + m * D_PAD + kbA;
  const unsigned short* bp = B + m * D_PAD + kbB;
#pragma unroll 1
  for (int k0 = 0; k0 < D_PAD; k0 += 32) {
    ABfrag a[BM], b[BN];
#pragma unroll
    for (int im = 0; im < BM; ++im) {
      const unsigned short* p = ap + im * 16 * D_PAD + k0;
      a[im].q[0] = *(const uint4*)(p);        // K = kbA .. kbA+7
      a[im].q[1] = *(const uint4*)(p + 16);   // K = kbA+16 .. kbA+23
    }
#pragma unroll
    for (int in = 0; in < BN; ++in) {
      const unsigned short* p = bp + in * 16 * D_PAD + k0;
      b[in].q[0] = *(const uint4*)(p);        // K = kbB .. kbB+7
      b[in].q[1] = *(const uint4*)(p + 16);   // K = kbB+16 .. kbB+23
    }
#pragma unroll
    for (int im = 0; im < BM; ++im)
#pragma unroll
      for (int in = 0; in < BN; ++in)
        acc[im][in] = __builtin_amdgcn_wmma_f32_16x16x32_bf16(
            /*neg_a=*/false, a[im].v, /*neg_b=*/false, b[in].v,
            /*c_mod=*/(short)0, acc[im][in],
            /*reuse_a=*/false, /*reuse_b=*/false);
  }
}

// GEMM1: Sbf[i,j] = bf16( sum_d E[i,d]*W[j,d] + b_fast[j] ), padded cols -> 0
__global__ void __launch_bounds__(32)
__attribute__((amdgpu_waves_per_eu(1)))
gemm1_kernel(const unsigned short* __restrict__ Ebf,
             const unsigned short* __restrict__ Wbf,
             const float* __restrict__ b_fast,
             unsigned short* __restrict__ Sbf) {
  int ct = blockIdx.x;   // 0..24 (64-col blocks over D_PAD)
  int rt = blockIdx.y;   // 0..24 (64-row blocks over K_TOP)
  v8f acc[BM][BN] = {};
  wmma_block(Ebf + rt * 64 * D_PAD, Wbf + ct * 64 * D_PAD, acc);
  int lane = threadIdx.x & 31;
  int nl = lane & 15;
  int mb = (lane >> 4) << 3;
#pragma unroll
  for (int in = 0; in < BN; ++in) {
    int n = ct * 64 + in * 16 + nl;
    float bias = (n < D_EMB) ? b_fast[n] : 0.0f;
#pragma unroll
    for (int im = 0; im < BM; ++im) {
#pragma unroll
      for (int r = 0; r < 8; ++r) {
        int i = rt * 64 + im * 16 + mb + r;
        Sbf[i * D_PAD + n] = f2bf(acc[im][in][r] + bias);  // pad cols -> 0
      }
    }
  }
}

// GEMM2 + fused coarse-score epilogue:
// F[i,j] = ts[i]+ts[j] + sum_d S[i,d]*E[j,d] + bucket[dist(i-j)] + mask
__global__ void __launch_bounds__(32)
__attribute__((amdgpu_waves_per_eu(1)))
gemm2_kernel(const unsigned short* __restrict__ Sbf,
             const unsigned short* __restrict__ Ebf,
             const float* __restrict__ ts,
             const float* __restrict__ bucket_scores,
             float* __restrict__ F) {
  int ct = blockIdx.x;   // 0..24 (64-col blocks over K_TOP)
  int rt = blockIdx.y;   // 0..24 (64-row blocks)
  v8f acc[BM][BN] = {};
  wmma_block(Sbf + rt * 64 * D_PAD, Ebf + ct * 64 * D_PAD, acc);
  int lane = threadIdx.x & 31;
  int nl = lane & 15;
  int mb = (lane >> 4) << 3;
#pragma unroll
  for (int in = 0; in < BN; ++in) {
    int j = ct * 64 + in * 16 + nl;
    float tsj = ts[j];
#pragma unroll
    for (int im = 0; im < BM; ++im) {
#pragma unroll
      for (int r = 0; r < 8; ++r) {
        int i = rt * 64 + im * 16 + mb + r;
        int off = i - j;
        int d = off > 0 ? off : 0;
        int idx = (d <= 4) ? d : min((31 - __clz(d)) + 3, 9);
        float f = ts[i] + tsj + acc[im][in][r] + bucket_scores[idx]
                  + ((off >= 1) ? 0.0f : NEG_BIG);
        F[i * K_TOP + j] = f;
      }
    }
  }
}

// ---------------------------------------------------------------------------
// 6) Per-row top-50: one wave per row, row cached in LDS, iterative argmax
//    with lowest-index tie-break (matches lax.top_k).
// ---------------------------------------------------------------------------
__global__ void topk_kernel(const float* __restrict__ F,
                            float* __restrict__ out_idx,
                            float* __restrict__ out_val) {
  __shared__ float row[K_TOP];
  int r = blockIdx.x;
  int lane = threadIdx.x;  // block = 32
  for (int j = lane; j < K_TOP; j += 32) row[j] = F[r * K_TOP + j];
  __syncthreads();
  for (int a = 0; a < A_ANT; ++a) {
    float best = -3.4e38f;
    int bidx = 0x7fffffff;
    for (int j = lane; j < K_TOP; j += 32) {
      float v = row[j];
      if (v > best || (v == best && j < bidx)) { best = v; bidx = j; }
    }
    for (int off = 16; off > 0; off >>= 1) {
      float ov = __shfl_xor(best, off, 32);
      int   oi = __shfl_xor(bidx, off, 32);
      if (ov > best || (ov == best && oi < bidx)) { best = ov; bidx = oi; }
    }
    if (lane == 0) {
      out_idx[r * A_ANT + a] = (float)bidx;
      out_val[r * A_ANT + a] = best;
      row[bidx] = -3.4e38f;
    }
    __syncthreads();
  }
}

// ---------------------------------------------------------------------------
extern "C" void kernel_launch(void* const* d_in, const int* in_sizes, int n_in,
                              void* d_out, int out_size, void* d_ws, size_t ws_size,
                              hipStream_t stream) {
  const int*   starts   = (const int*)  d_in[0];
  const int*   ends     = (const int*)  d_in[1];
  const float* scores   = (const float*)d_in[2];
  const float* emb      = (const float*)d_in[3];
  const float* W_fast   = (const float*)d_in[4];
  const float* b_fast   = (const float*)d_in[5];
  const float* emb_dist = (const float*)d_in[6];
  const float* W_dist   = (const float*)d_in[7];
  const float* b_dist   = (const float*)d_in[8];

  char* ws = (char*)d_ws;
  size_t off = 0;
  auto alloc = [&](size_t bytes) -> char* {
    char* p = ws + off;
    off = (off + bytes + 255) & ~(size_t)255;
    return p;
  };
  int*   order   = (int*)  alloc(C_CAND * 4);
  int*   s_sort  = (int*)  alloc(C_CAND * 4);
  int*   e_sort  = (int*)  alloc(C_CAND * 4);
  int*   selfl   = (int*)  alloc(C_CAND * 4);
  int*   n_sel   = (int*)  alloc(256);
  int*   cbr     = (int*)  alloc(K_TOP * 4);
  int*   top     = (int*)  alloc(K_TOP * 4);
  float* ts_top  = (float*)alloc(K_TOP * 4);
  float* buckets = (float*)alloc(256);
  unsigned short* Ebf = (unsigned short*)alloc((size_t)K_TOP * D_PAD * 2);
  unsigned short* Wbf = (unsigned short*)alloc((size_t)D_PAD * D_PAD * 2);
  unsigned short* Sbf = (unsigned short*)alloc((size_t)K_TOP * D_PAD * 2);
  float* F = (float*)alloc((size_t)K_TOP * K_TOP * 4);
  (void)ws_size; (void)in_sizes; (void)n_in; (void)out_size;

  float* out_top    = (float*)d_out;                       // [K_TOP]
  float* out_antidx = out_top + K_TOP;                     // [K_TOP*A_ANT]
  float* out_antval = out_antidx + (size_t)K_TOP * A_ANT;  // [K_TOP*A_ANT]

  const int blocksC = (C_CAND + 255) / 256;

  rank_scores_kernel<<<blocksC, 256, 0, stream>>>(scores, order);
  gather_se_kernel<<<blocksC, 256, 0, stream>>>(order, starts, ends, s_sort, e_sort);
  greedy_select_kernel<<<1, 32, 0, stream>>>(s_sort, e_sort, selfl, n_sel);
  rank_selected_kernel<<<blocksC, 256, 0, stream>>>(s_sort, e_sort, selfl, order, cbr);
  fill_top_kernel<<<(K_TOP + 255) / 256, 256, 0, stream>>>(cbr, n_sel, top, out_top);

  gather_emb_kernel<<<(K_TOP * D_PAD + 255) / 256, 256, 0, stream>>>(
      top, emb, scores, Ebf, ts_top);
  conv_w_kernel<<<(D_PAD * D_PAD + 255) / 256, 256, 0, stream>>>(W_fast, Wbf);
  bucket_scores_kernel<<<1, 32, 0, stream>>>(emb_dist, W_dist, b_dist, buckets);

  gemm1_kernel<<<dim3(D_PAD / 64, K_TOP / 64), 32, 0, stream>>>(Ebf, Wbf, b_fast, Sbf);
  gemm2_kernel<<<dim3(K_TOP / 64, K_TOP / 64), 32, 0, stream>>>(Sbf, Ebf, ts_top, buckets, F);

  topk_kernel<<<K_TOP, 32, 0, stream>>>(F, out_antidx, out_antval);
}
